// DCN_326417515094
// MI455X (gfx1250) — matrix operand order repaired and checked by reading, not compiled, not run on hardware
//
#include <hip/hip_runtime.h>

#define BATCH 8
#define CH 64
#define HH 128
#define WW 128
#define OCH 64
#define OFFC 18
#define KK9 9
#define HW (HH*WW)

typedef __attribute__((ext_vector_type(16))) __bf16 v16bf;
typedef __attribute__((ext_vector_type(8)))  __bf16 v8bf;
typedef __attribute__((ext_vector_type(8)))  float  v8f;

// B-matrix 16-bit 32x16 layout: lane group g holds K = 8g..8g+7 (j<8) and 16+8g..16+8g+7 (j>=8)
__device__ __forceinline__ int klocB(int j, int g) { return (j < 8) ? (8 * g + j) : (16 + 8 * g + (j - 8)); }

__device__ __forceinline__ v16bf cat16(v8bf lo, v8bf hi) {
    return __builtin_shufflevector(lo, hi, 0, 1, 2, 3, 4, 5, 6, 7,
                                           8, 9, 10, 11, 12, 13, 14, 15);
}

__global__ void dcn_zero(float* ws) {
    if (threadIdx.x == 0) ws[0] = 0.0f;
}

// ---------------------------------------------------------------------------
// Kernel 2: 3x3 offset conv as implicit GEMM with bf16 WMMA.
// Block = one (b,h) row; 8 waves x 16 pixels. M=pixels, N=offset channels,
// K = C*9 = 576, processed as two c-half passes of 9 kk-chunks (K=32 each).
// fea halo tile staged in LDS as [row][x][c] bf16 so A-fragments are two
// ds_load_b128 of 8 consecutive channels (matches 16-bit A VGPR layout).
// ---------------------------------------------------------------------------
__global__ __launch_bounds__(256)
void dcn_offset(const float* __restrict__ fea, const float* __restrict__ w_off,
                const float* __restrict__ b_off, float* __restrict__ ws)
{
    __shared__ __align__(16) __bf16 lA[3 * 132 * 32];   // 25.3 KB fea tile (one c-half)
    __shared__ __align__(16) __bf16 lB[9 * 2 * 32 * 16]; // 18 KB w_off fragments
    __shared__ float red[256];

    const int tid = threadIdx.x;
    const int bb  = blockIdx.x / HH;
    const int h   = blockIdx.x % HH;
    const int lane = tid & 31;
    const int wave = tid >> 5;
    const int g    = lane >> 4;
    const int w0   = wave * 16;
    const int pw   = w0 + (lane & 15);

    v8f acc0 = {};
    v8f acc1 = {};

    for (int pass = 0; pass < 2; ++pass) {
        const int cbase = 32 * pass;
        if (pass) __syncthreads();

        // Stage fea tile: lA[yy][xx][cc] = fea[bb][cbase+cc][h+yy-1][xx-1], 0-padded.
        // idx decoded x-fastest so global reads are coalesced along W.
        for (int idx = tid; idx < 3 * 132 * 32; idx += 256) {
            int xx = idx % 132;
            int r  = idx / 132;
            int yy = r % 3;
            int cc = r / 3;
            int y = h + yy - 1;
            int x = xx - 1;
            float v = 0.0f;
            if (y >= 0 && y < HH && x >= 0 && x < WW)
                v = fea[(((size_t)bb * CH + cbase + cc) * HH + y) * WW + x];
            lA[(yy * 132 + xx) * 32 + cc] = (__bf16)v;
        }
        // Stage w_off fragments for this c-half.
        for (int idx = tid; idx < 9 * 2 * 32 * 16; idx += 256) {
            int j  = idx & 15;
            int ln = (idx >> 4) & 31;
            int nt = (idx >> 9) & 1;
            int kk = idx >> 10;
            int c  = cbase + klocB(j, ln >> 4);
            int o  = nt * 16 + (ln & 15);
            float v = (o < OFFC) ? w_off[(o * CH + c) * KK9 + kk] : 0.0f;
            lB[idx] = (__bf16)v;
        }
        __syncthreads();

#pragma unroll
        for (int kk = 0; kk < 9; ++kk) {
            const int dy = kk / 3;            // tile row index (already -1 shifted)
            const int dx = kk % 3;            // tile x shift (xx = pw + dx)
            const int abase = (dy * 132 + pw + dx) * 32;
            v8bf a0 = *(const v8bf*)&lA[abase + 8 * g];
            v8bf a1 = *(const v8bf*)&lA[abase + 16 + 8 * g];
            v16bf a = cat16(a0, a1);
            v16bf b0 = *(const v16bf*)&lB[((kk * 2 + 0) * 32 + lane) * 16];
            v16bf b1 = *(const v16bf*)&lB[((kk * 2 + 1) * 32 + lane) * 16];
            acc0 = __builtin_amdgcn_wmma_f32_16x16x32_bf16(false, a, false, b0,
                                                           (short)0, acc0, false, false);
            acc1 = __builtin_amdgcn_wmma_f32_16x16x32_bf16(false, a, false, b1,
                                                           (short)0, acc1, false, false);
        }
    }

    // Epilogue: bias, clip, store offsets, accumulate |offset| for the mean.
    float* wsoff = ws + 16;
    float s = 0.0f;
#pragma unroll
    for (int r = 0; r < 8; ++r) {
        int pix = w0 + r + 8 * g;
        {
            int o = lane & 15;
            float v = acc0[r] + b_off[o];
            v = fminf(fmaxf(v, -200.0f), 200.0f);
            wsoff[((size_t)(bb * OFFC + o) * HH + h) * WW + pix] = v;
            s += fabsf(v);
        }
        {
            int o = 16 + (lane & 15);
            if (o < OFFC) {
                float v = acc1[r] + b_off[o];
                v = fminf(fmaxf(v, -200.0f), 200.0f);
                wsoff[((size_t)(bb * OFFC + o) * HH + h) * WW + pix] = v;
                s += fabsf(v);
            }
        }
    }
    red[tid] = s;
    __syncthreads();
    for (int st = 128; st > 0; st >>= 1) {
        if (tid < st) red[tid] += red[tid + st];
        __syncthreads();
    }
    if (tid == 0) atomicAdd(&ws[0], red[0]);
}

// ---------------------------------------------------------------------------
// Kernel 3: bilinear gather + main contraction (M=16 pixels, N=64 out chans,
// K=576 as two c-half passes of 9 kk-chunks). Branchless gather: corner
// validity folded into bilinear weights, indices clamped, 4 unconditional
// L2-resident loads + FMAs per A element.
// ---------------------------------------------------------------------------
__global__ __launch_bounds__(256)
void dcn_main(const float* __restrict__ input, const float* __restrict__ w_dcn,
              const float* __restrict__ b_dcn, const float* __restrict__ ws,
              float* __restrict__ out)
{
    __shared__ __align__(16) __bf16 lB[9 * 4 * 32 * 16]; // 36 KB per c-half pass

    const int tid = threadIdx.x;
    const int bb  = blockIdx.x / HH;
    const int h   = blockIdx.x % HH;
    const int lane = tid & 31;
    const int wave = tid >> 5;
    const int g    = lane >> 4;
    const int w0   = wave * 16;
    const int pw   = w0 + (lane & 15);

    const float* inb   = input + (size_t)bb * CH * HW;
    const float* wsoff = ws + 16;
    const float mean = ws[0] * (1.0f / (float)(BATCH * OFFC * HH * WW));
    const bool  thr  = (mean > 150.0f);

    v8f acc[4] = {};

    for (int pass = 0; pass < 2; ++pass) {
        const int cbase = 32 * pass;
        if (pass) __syncthreads();

        // Stage this c-half of w_dcn into LDS in B-fragment order.
        for (int idx = tid; idx < 9 * 4 * 32 * 16; idx += 256) {
            int j  = idx & 15;
            int ln = (idx >> 4) & 31;
            int mt = (idx >> 9) & 3;
            int kk = idx >> 11;
            int c  = cbase + klocB(j, ln >> 4);
            int o  = mt * 16 + (ln & 15);
            lB[idx] = (__bf16)w_dcn[(o * CH + c) * KK9 + kk];
        }
        __syncthreads();

        for (int kk = 0; kk < 9; ++kk) {
            float oy = wsoff[((size_t)(bb * OFFC + 2 * kk + 0) * HH + h) * WW + pw];
            float ox = wsoff[((size_t)(bb * OFFC + 2 * kk + 1) * HH + h) * WW + pw];
            if (thr) {
                if (oy >= 150.0f) oy = 0.001f;
                if (ox >= 150.0f) ox = 0.001f;
            }
            float py = oy + (float)(h + kk / 3 - 1);
            float px = ox + (float)(pw + kk % 3 - 1);
            float fy = floorf(py), fx = floorf(px);
            float wy = py - fy,    wx = px - fx;
            int y0 = (int)fy, x0 = (int)fx;
            int y1 = y0 + 1,  x1 = x0 + 1;
            // validity folded into weights (branchless)
            float my0 = (y0 >= 0 && y0 < HH) ? 1.0f : 0.0f;
            float my1 = (y1 >= 0 && y1 < HH) ? 1.0f : 0.0f;
            float mx0 = (x0 >= 0 && x0 < WW) ? 1.0f : 0.0f;
            float mx1 = (x1 >= 0 && x1 < WW) ? 1.0f : 0.0f;
            float w00 = (1.0f - wy) * (1.0f - wx) * my0 * mx0;
            float w01 = (1.0f - wy) * wx          * my0 * mx1;
            float w10 = wy * (1.0f - wx)          * my1 * mx0;
            float w11 = wy * wx                   * my1 * mx1;
            int y0c = min(max(y0, 0), HH - 1), y1c = min(max(y1, 0), HH - 1);
            int x0c = min(max(x0, 0), WW - 1), x1c = min(max(x1, 0), WW - 1);
            int i00 = y0c * WW + x0c, i01 = y0c * WW + x1c;
            int i10 = y1c * WW + x0c, i11 = y1c * WW + x1c;

            const float* pbase = inb + (size_t)cbase * HW;
            v16bf a;
#pragma unroll
            for (int j = 0; j < 16; ++j) {
                int cc = (j < 8) ? (8 * g + j) : (16 + 8 * g + (j - 8));
                const float* p = pbase + (size_t)cc * HW;
                float s = fmaf(w00, p[i00],
                          fmaf(w01, p[i01],
                          fmaf(w10, p[i10], w11 * p[i11])));
                a[j] = (__bf16)s;
            }
#pragma unroll
            for (int mt = 0; mt < 4; ++mt) {
                v16bf bfr = *(const v16bf*)&lB[((kk * 4 + mt) * 32 + lane) * 16];
                acc[mt] = __builtin_amdgcn_wmma_f32_16x16x32_bf16(false, a, false, bfr,
                                                                  (short)0, acc[mt],
                                                                  false, false);
            }
        }
    }

    // Epilogue: D layout: VGPR r, group g -> pixel = w0 + r + 8g, N = lane&15.
#pragma unroll
    for (int mt = 0; mt < 4; ++mt) {
        int o = mt * 16 + (lane & 15);
        float bias = b_dcn[o];
#pragma unroll
        for (int r = 0; r < 8; ++r) {
            int pix = w0 + r + 8 * g;
            out[((size_t)(bb * OCH + o) * HH + h) * WW + pix] = acc[mt][r] + bias;
        }
    }
}

extern "C" void kernel_launch(void* const* d_in, const int* in_sizes, int n_in,
                              void* d_out, int out_size, void* d_ws, size_t ws_size,
                              hipStream_t stream) {
    const float* input = (const float*)d_in[0];
    const float* fea   = (const float*)d_in[1];
    const float* w_off = (const float*)d_in[2];
    const float* b_off = (const float*)d_in[3];
    const float* w_dcn = (const float*)d_in[4];
    const float* b_dcn = (const float*)d_in[5];
    float* out = (float*)d_out;
    float* ws  = (float*)d_ws; // [0]=sum|offset|, [16..] = offsets (B*18*H*W floats)

    dcn_zero<<<1, 32, 0, stream>>>(ws);
    dcn_offset<<<BATCH * HH, 256, 0, stream>>>(fea, w_off, b_off, ws);
    dcn_main<<<BATCH * HH, 256, 0, stream>>>(input, w_dcn, b_dcn, ws, out);
}
